// Probability_66048007078103
// MI455X (gfx1250) — compile-verified, matrix operand order checked by reading, not run on hardware
//
#include <hip/hip_runtime.h>

// ---------------------------------------------------------------------------
// Probability histogram kernel for MI455X (gfx1250, wave32).
//
// inputs: [B=64, P=4096, 16, 16] f32 (binary 0/1)
// output: [B, 257] f32 histogram of per-patch sums, divided by P.
//
// Strategy (memory-bound: 256 MiB @ 23.3 TB/s ~ 11.5 us floor):
//   * coalesced float4 global loads -> f16 -> LDS (bank-conflict-free padding)
//   * per-wave: 8x v_wmma_f32_16x16x32_f16 with B = ones reduces 16 patches
//   * lane 0 / lane 16 hold the 16 patch sums -> ds_add_u32 LDS histogram
//   * block merges into d_out with global_atomic_add_f32 (exact, order-free)
// ---------------------------------------------------------------------------

typedef __attribute__((ext_vector_type(16))) _Float16 v16h;
typedef __attribute__((ext_vector_type(8)))  _Float16 v8h;
typedef __attribute__((ext_vector_type(4)))  _Float16 v4h;
typedef __attribute__((ext_vector_type(8)))  float    v8f;

#define BATCH        64
#define PATCHES      4096
#define PATCH_ELEMS  256          // 16x16
#define NBINS        257          // s*s + 1
#define TILE_PATCHES 64           // patches per block
#define THREADS      128          // 4 waves of 32
#define STRIDE_H     264          // 256 f16 elems + 8 halfs pad -> 528 B/patch
                                  // 528 B = 132 dwords -> bank advance 4/row:
                                  // 16-lane ds_read_b128 groups are conflict-free

__global__ void zero_out_kernel(float* __restrict__ out, int n) {
  int i = blockIdx.x * blockDim.x + threadIdx.x;
  if (i < n) out[i] = 0.0f;
}

__global__ __launch_bounds__(THREADS)
void patch_hist_kernel(const float* __restrict__ in, float* __restrict__ out) {
  __shared__ __align__(16) _Float16 stage[TILE_PATCHES * STRIDE_H]; // 33792 B
  __shared__ unsigned int hist[NBINS];                              // 1028 B

  const int tid  = threadIdx.x;
  const int tile = blockIdx.x;   // 0..63 : which 64-patch tile
  const int b    = blockIdx.y;   // 0..63 : batch

  for (int t = tid; t < NBINS; t += THREADS) hist[t] = 0u;

  // ---- Stage: coalesced b128 global loads, cvt f32->f16, store to LDS ----
  // Tile = 64 patches * 256 f32 = 64 KiB = 4096 float4; 32 float4 per thread.
  const float4* __restrict__ gsrc =
      (const float4*)(in + ((size_t)b * PATCHES + (size_t)tile * TILE_PATCHES)
                               * PATCH_ELEMS);
#pragma unroll
  for (int k = 0; k < 32; ++k) {
    const int i  = tid + k * THREADS;  // float4 index in tile (0..4095)
    const float4 v = gsrc[i];
    const int p  = i >> 6;             // 64 float4 per patch
    const int e4 = i & 63;
    v4h h;
    h[0] = (_Float16)v.x;
    h[1] = (_Float16)v.y;
    h[2] = (_Float16)v.z;
    h[3] = (_Float16)v.w;
    *(v4h*)&stage[p * STRIDE_H + e4 * 4] = h;   // 8 B aligned ds_store
  }
  __syncthreads();

  // ---- Reduce: each wave sums its 16 patches with 8 chained WMMAs ----
  // A (16x32 f16): row m = patch (pbase+m), K = 32-element slab c.
  //   lanes 0-15 : VGPR0..3 = K[c*32 + 0..7],  VGPR4..7 = K[c*32 + 16..23]
  //   lanes 16-31: VGPR0..3 = K[c*32 + 8..15], VGPR4..7 = K[c*32 + 24..31]
  // B = all ones  =>  D[m][n] = sum_k A[m][k]  (replicated over n).
  const int wave  = tid >> 5;
  const int lane  = tid & 31;
  const int half  = lane >> 4;   // 0: K lo-octet, 1: K hi-octet
  const int lrow  = lane & 15;   // patch row within wave's 16-patch tile
  const int pbase = wave * 16;

  v16h bones;
#pragma unroll
  for (int i = 0; i < 16; ++i) bones[i] = (_Float16)1.0f;

  v8f acc = {};
  const _Float16* __restrict__ rowp =
      &stage[(pbase + lrow) * STRIDE_H + half * 8];
#pragma unroll
  for (int c = 0; c < 8; ++c) {
    const v8h lo = *(const v8h*)(rowp + c * 32);        // ds_read_b128
    const v8h hi = *(const v8h*)(rowp + c * 32 + 16);   // ds_read_b128
    const v16h a = __builtin_shufflevector(
        lo, hi, 0, 1, 2, 3, 4, 5, 6, 7, 8, 9, 10, 11, 12, 13, 14, 15);
    acc = __builtin_amdgcn_wmma_f32_16x16x32_f16(
        /*neg_a=*/false, a, /*neg_b=*/false, bones,
        /*c_mod=*/(short)0, acc, /*reuse_a=*/false, /*reuse_b=*/false);
  }

  // C/D layout: lane 0 VGPR v = patch pbase+v ; lane 16 VGPR v = patch pbase+8+v.
  if ((lane & 15) == 0) {
    const int moff = half ? 8 : 0;
    (void)moff; // sums identical across N; lane 0 -> M 0..7, lane 16 -> M 8..15
#pragma unroll
    for (int v = 0; v < 8; ++v) {
      int bin = (int)(acc[v] + 0.5f);         // sums are exact integers
      bin = bin < 0 ? 0 : (bin > 256 ? 256 : bin);
      atomicAdd(&hist[bin], 1u);              // ds_add_u32
    }
  }
  __syncthreads();

  // ---- Merge block histogram into global output ----
  // Each contribution is k * 2^-12 with totals <= 1.0: every partial sum is an
  // exact f32, so atomic order does not affect the result (deterministic).
  for (int t = tid; t < NBINS; t += THREADS) {
    const float val = (float)hist[t] * (1.0f / (float)PATCHES);
    atomicAdd(&out[(size_t)b * NBINS + t], val);  // global_atomic_add_f32
  }
}

extern "C" void kernel_launch(void* const* d_in, const int* in_sizes, int n_in,
                              void* d_out, int out_size, void* d_ws, size_t ws_size,
                              hipStream_t stream) {
  const float* in  = (const float*)d_in[0];
  float*       out = (float*)d_out;

  // d_out is poisoned by the harness; zero it first (stream-ordered).
  zero_out_kernel<<<(out_size + 255) / 256, 256, 0, stream>>>(out, out_size);

  dim3 grid(PATCHES / TILE_PATCHES, BATCH);  // (64, 64) = 4096 blocks
  patch_hist_kernel<<<grid, THREADS, 0, stream>>>(in, out);
}
